// MessagePassingLayer_83348135346321
// MI455X (gfx1250) — matrix operand order repaired and checked by reading, not compiled, not run on hardware
//
#include <hip/hip_runtime.h>
#include <hip/hip_bf16.h>
#include <math.h>

#define N_NODES   100000
#define N_EDGES   1200000
#define HIDDEN    64
#define EDGE_DIM  16
#define NODE_TILES (N_NODES / 16)   // 6250, exact

typedef __attribute__((ext_vector_type(16))) __bf16      v16bf;
typedef __attribute__((ext_vector_type(8)))  float       v8f;
typedef __attribute__((ext_vector_type(4)))  unsigned int v4u;

union Frag {
  unsigned short u[16];
  v16bf v;
  v4u   q[2];
};

__device__ __forceinline__ unsigned short f2bf(float f) {
  unsigned int x = __float_as_uint(f);
  unsigned int r = x + 0x7FFFu + ((x >> 16) & 1u);   // round-to-nearest-even
  return (unsigned short)(r >> 16);
}

__device__ __forceinline__ float gelu_tanh(float x) {
  const float c = 0.7978845608028654f;               // sqrt(2/pi)
  float t = tanhf(c * (x + 0.044715f * x * x * x));
  return 0.5f * x * (1.0f + t);
}

// Pack a fp32 weight [kreal x 64] into bf16 B-fragments, padded to ksteps*32 in K.
// Fragment f = s*4 + n; lane L holds 16 bf16: K = s*32 + (j>=8 ? 16 : 0) + (L>=16 ? 8 : 0) + (j&7),
// col = n*16 + (L & 15). One 32B contiguous chunk per (f, lane).
__global__ void pack_weight_kernel(const float* __restrict__ W,
                                   unsigned short* __restrict__ Wp,
                                   int ksteps, int kreal) {
  int total = ksteps * 4 * 32 * 16;
  for (int i = blockIdx.x * blockDim.x + threadIdx.x; i < total;
       i += gridDim.x * blockDim.x) {
    int j = i & 15;
    int L = (i >> 4) & 31;
    int f = i >> 9;
    int s = f >> 2, n = f & 3;
    int K = s * 32 + ((j >> 3) << 4) + ((L >> 4) << 3) + (j & 7);
    int col = n * 16 + (L & 15);
    float v = (K < kreal) ? W[K * 64 + col] : 0.0f;
    Wp[i] = f2bf(v);
  }
}

// Edge MLP + scatter-add. 256 threads = 8 waves; each wave owns 16 edges.
__global__ __launch_bounds__(256)
void edge_kernel(const float* __restrict__ nodef, const float* __restrict__ edgef,
                 const long long* __restrict__ eidx,
                 const unsigned short* __restrict__ W1p, const float* __restrict__ b1,
                 const unsigned short* __restrict__ W2p, const float* __restrict__ b2,
                 float* __restrict__ agg) {
  __shared__ unsigned short sA[8][16 * 96];   // concat(src_feat, edge_feat) padded to 96
  __shared__ unsigned short sH[8][16 * 64];   // gelu(layer1) for layer2's A operand

  const int wave = threadIdx.x >> 5;
  const int lane = threadIdx.x & 31;
  const int tile = blockIdx.x * 8 + wave;
  const int ebase = tile * 16;
  const long long* __restrict__ src = eidx;
  const long long* __restrict__ dst = eidx + N_EDGES;

  // Stage 16 x 96 bf16 tile (gather from node_features via L2-resident table).
  for (int i = lane; i < 16 * 96; i += 32) {
    int row = i / 96, k = i - row * 96;
    int e = ebase + row;
    float v = 0.0f;
    if (k < 64)       v = nodef[(size_t)src[e] * 64 + k];
    else if (k < 80)  v = edgef[(size_t)e * 16 + (k - 64)];
    sA[wave][i] = f2bf(v);
  }
  __syncthreads();

  const int rowA = lane & 15;
  const int hi   = lane >> 4;      // A-layout: hi half holds K+8 / K+24 slices
  const int koff = hi * 8;
  const int colB = lane & 15;

  const v8f vzero = {0.f, 0.f, 0.f, 0.f, 0.f, 0.f, 0.f, 0.f};
  v8f acc[4];
  for (int n = 0; n < 4; ++n) acc[n] = vzero;

  // Layer 1: [16x96] x [96x64], 3 K-steps x 4 N-tiles
  for (int s = 0; s < 3; ++s) {
    Frag a;
    a.q[0] = *(const v4u*)&sA[wave][rowA * 96 + s * 32 + koff];
    a.q[1] = *(const v4u*)&sA[wave][rowA * 96 + s * 32 + 16 + koff];
    for (int n = 0; n < 4; ++n) {
      Frag b;
      const v4u* wp = (const v4u*)&W1p[(((s * 4 + n) * 32) + lane) * 16];
      b.q[0] = wp[0];
      b.q[1] = wp[1];
      acc[n] = __builtin_amdgcn_wmma_f32_16x16x32_bf16(
          false, a.v, false, b.v, (short)0, acc[n], false, false);
    }
  }

  // Bias + GELU (f32), stash as bf16 A-operand for layer 2.
  for (int n = 0; n < 4; ++n) {
    int col = n * 16 + colB;
    float bias = b1[col];
    for (int r = 0; r < 8; ++r) {
      float g = gelu_tanh(acc[n][r] + bias);
      sH[wave][(r + hi * 8) * 64 + col] = f2bf(g);
    }
  }
  __syncthreads();

  // Layer 2: [16x64] x [64x64], 2 K-steps x 4 N-tiles
  v8f acc2[4];
  for (int n = 0; n < 4; ++n) acc2[n] = vzero;
  for (int s = 0; s < 2; ++s) {
    Frag a;
    a.q[0] = *(const v4u*)&sH[wave][rowA * 64 + s * 32 + koff];
    a.q[1] = *(const v4u*)&sH[wave][rowA * 64 + s * 32 + 16 + koff];
    for (int n = 0; n < 4; ++n) {
      Frag b;
      const v4u* wp = (const v4u*)&W2p[(((s * 4 + n) * 32) + lane) * 16];
      b.q[0] = wp[0];
      b.q[1] = wp[1];
      acc2[n] = __builtin_amdgcn_wmma_f32_16x16x32_bf16(
          false, a.v, false, b.v, (short)0, acc2[n], false, false);
    }
  }

  // Scatter-add into agg (L2-resident, 25.6 MB).
  long long drow[8];
  for (int r = 0; r < 8; ++r) drow[r] = dst[ebase + hi * 8 + r];
  for (int n = 0; n < 4; ++n) {
    int col = n * 16 + colB;
    float bias = b2[col];
    for (int r = 0; r < 8; ++r) {
      atomicAdd(&agg[(size_t)drow[r] * 64 + col], acc2[n][r] + bias);
    }
  }
}

// Node update: dense + LayerNorm + GELU + residual. 8 waves x 16 nodes.
__global__ __launch_bounds__(256)
void node_kernel(const float* __restrict__ nodef, const float* __restrict__ agg,
                 const unsigned short* __restrict__ W3p, const float* __restrict__ b3,
                 const float* __restrict__ ln_scale, const float* __restrict__ ln_bias,
                 float* __restrict__ out) {
  __shared__ unsigned short sA[8][16 * 128];

  const int wave = threadIdx.x >> 5;
  const int lane = threadIdx.x & 31;
  const int tile = blockIdx.x * 8 + wave;
  const bool valid = tile < NODE_TILES;
  const int t = valid ? tile : (NODE_TILES - 1);
  const int nbase = t * 16;

  for (int i = lane; i < 16 * 128; i += 32) {
    int row = i >> 7, k = i & 127;
    size_t node = (size_t)(nbase + row);
    float v = (k < 64) ? nodef[node * 64 + k] : agg[node * 64 + (k - 64)];
    sA[wave][i] = f2bf(v);
  }
  __syncthreads();

  const int rowA = lane & 15;
  const int hi   = lane >> 4;
  const int koff = hi * 8;
  const int colB = lane & 15;

  const v8f vzero = {0.f, 0.f, 0.f, 0.f, 0.f, 0.f, 0.f, 0.f};
  v8f acc[4];
  for (int n = 0; n < 4; ++n) acc[n] = vzero;

  // [16x128] x [128x64]: 4 K-steps x 4 N-tiles
  for (int s = 0; s < 4; ++s) {
    Frag a;
    a.q[0] = *(const v4u*)&sA[wave][rowA * 128 + s * 32 + koff];
    a.q[1] = *(const v4u*)&sA[wave][rowA * 128 + s * 32 + 16 + koff];
    for (int n = 0; n < 4; ++n) {
      Frag b;
      const v4u* wp = (const v4u*)&W3p[(((s * 4 + n) * 32) + lane) * 16];
      b.q[0] = wp[0];
      b.q[1] = wp[1];
      acc[n] = __builtin_amdgcn_wmma_f32_16x16x32_bf16(
          false, a.v, false, b.v, (short)0, acc[n], false, false);
    }
  }

  // Bias, then LayerNorm stats. Row m = hi*8+r: its 64 cols live across the
  // 16 lanes of this half-wave (4 tiles x colB) -> shfl_xor reduce over 16.
  float x[4][8];
  float s1[8], s2[8];
  for (int r = 0; r < 8; ++r) { s1[r] = 0.f; s2[r] = 0.f; }
  for (int n = 0; n < 4; ++n) {
    float bias = b3[n * 16 + colB];
    for (int r = 0; r < 8; ++r) {
      float v = acc[n][r] + bias;
      x[n][r] = v;
      s1[r] += v;
      s2[r] += v * v;
    }
  }
  for (int m = 1; m <= 8; m <<= 1) {
    for (int r = 0; r < 8; ++r) {
      s1[r] += __shfl_xor(s1[r], m, 32);
      s2[r] += __shfl_xor(s2[r], m, 32);
    }
  }

  float mean[8], rstd[8];
  for (int r = 0; r < 8; ++r) {
    mean[r] = s1[r] * (1.0f / 64.0f);
    float var = s2[r] * (1.0f / 64.0f) - mean[r] * mean[r];
    rstd[r] = rsqrtf(var + 1e-6f);
  }

  for (int n = 0; n < 4; ++n) {
    int col = n * 16 + colB;
    float g = ln_scale[col];
    float bb = ln_bias[col];
    for (int r = 0; r < 8; ++r) {
      size_t node = (size_t)(nbase + hi * 8 + r);
      float y = (x[n][r] - mean[r]) * rstd[r] * g + bb;
      float o = gelu_tanh(y) + nodef[node * 64 + col];
      if (valid) out[node * 64 + col] = o;
    }
  }
}

extern "C" void kernel_launch(void* const* d_in, const int* in_sizes, int n_in,
                              void* d_out, int out_size, void* d_ws, size_t ws_size,
                              hipStream_t stream) {
  const float*     nodef = (const float*)d_in[0];
  const float*     edgef = (const float*)d_in[1];
  const long long* eidx  = (const long long*)d_in[2];
  const float*     W1    = (const float*)d_in[3];
  const float*     b1    = (const float*)d_in[4];
  const float*     W2    = (const float*)d_in[5];
  const float*     b2    = (const float*)d_in[6];
  const float*     W3    = (const float*)d_in[7];
  const float*     b3    = (const float*)d_in[8];
  const float*     lns   = (const float*)d_in[9];
  const float*     lnb   = (const float*)d_in[10];
  float* out = (float*)d_out;

  char* ws = (char*)d_ws;
  const size_t aggBytes = (size_t)N_NODES * HIDDEN * sizeof(float);   // 25.6 MB
  float* agg = (float*)ws;
  unsigned short* W1p = (unsigned short*)(ws + aggBytes);             // 3 ksteps
  unsigned short* W2p = W1p + 3 * 4 * 32 * 16;                        // 2 ksteps
  unsigned short* W3p = W2p + 2 * 4 * 32 * 16;                        // 4 ksteps

  hipMemsetAsync(agg, 0, aggBytes, stream);
  pack_weight_kernel<<<24, 256, 0, stream>>>(W1, W1p, 3, HIDDEN + EDGE_DIM);
  pack_weight_kernel<<<16, 256, 0, stream>>>(W2, W2p, 2, HIDDEN);
  pack_weight_kernel<<<32, 256, 0, stream>>>(W3, W3p, 4, 2 * HIDDEN);

  edge_kernel<<<N_EDGES / 128, 256, 0, stream>>>(nodef, edgef, eidx,
                                                 W1p, b1, W2p, b2, agg);
  node_kernel<<<(NODE_TILES + 7) / 8, 256, 0, stream>>>(nodef, agg, W3p, b3,
                                                        lns, lnb, out);
}
